// GAT_BGNN_51453708206731
// MI455X (gfx1250) — matrix operand order repaired and checked by reading, not compiled, not run on hardware
//
#include <hip/hip_runtime.h>
#include <hip/hip_bf16.h>

// ---------------- problem constants ----------------
#define NN    20000      // nodes
#define E0    160000     // raw edges
#define ET    180000     // edges + self loops
#define HID   256
#define HEADS 8
#define F1    2048       // HEADS*HID
#define KT2   64         // F1/32 k-tiles for WMMA gemm2

typedef __attribute__((ext_vector_type(16))) _Float16 v16h;
typedef __attribute__((ext_vector_type(8)))  float    v8f;

// edge_index is int64 in the reference; harness convention passes integer
// tensors as 32-bit ints on device (values < 2^31 so lossless).
__device__ __forceinline__ void edge_sd(const int* __restrict__ ei, int e, int& s, int& d) {
    if (e < E0) { s = ei[e]; d = ei[E0 + e]; }
    else        { s = e - E0; d = e - E0; }
}

// monotone uint encoding of float for atomicMax-based segment max
__device__ __forceinline__ unsigned fenc(float f) {
    unsigned u = __float_as_uint(f);
    return (u & 0x80000000u) ? ~u : (u | 0x80000000u);
}
__device__ __forceinline__ float fdec(unsigned u) {
    return (u & 0x80000000u) ? __uint_as_float(u & 0x7FFFFFFFu) : __uint_as_float(~u);
}

// ---------------- K1: x @ W1 -> h1[N,2048]; per-node attention logits ----------------
__global__ void k_gemm1_att(const float* __restrict__ x, const float* __restrict__ W1,
                            const float* __restrict__ as1, const float* __restrict__ ad1,
                            float* __restrict__ h1, float* __restrict__ a_src,
                            float* __restrict__ a_dst) {
    int n = blockIdx.x;
    int t = threadIdx.x;           // 256 threads; thread t -> c = t, heads 0..7
    float xv[7];
#pragma unroll
    for (int k = 0; k < 7; ++k) xv[k] = x[n * 7 + k];
    float ps[HEADS], pd[HEADS];
#pragma unroll
    for (int j = 0; j < HEADS; ++j) {
        int col = j * 256 + t;
        float acc = 0.f;
#pragma unroll
        for (int k = 0; k < 7; ++k) acc += xv[k] * W1[k * F1 + col];
        h1[(size_t)n * F1 + col] = acc;
        ps[j] = acc * as1[col];    // att_src1[j][t]
        pd[j] = acc * ad1[col];
    }
    // wave-level tree reduction (wave32)
#pragma unroll
    for (int j = 0; j < HEADS; ++j)
        for (int m = 16; m >= 1; m >>= 1) {
            ps[j] += __shfl_xor(ps[j], m, 32);
            pd[j] += __shfl_xor(pd[j], m, 32);
        }
    __shared__ float sred[8][16];
    int w = t >> 5, lane = t & 31;
    if (lane == 0) {
#pragma unroll
        for (int j = 0; j < HEADS; ++j) { sred[w][j] = ps[j]; sred[w][8 + j] = pd[j]; }
    }
    __syncthreads();
    if (t < 16) {
        float s = 0.f;
#pragma unroll
        for (int w2 = 0; w2 < 8; ++w2) s += sred[w2][t];
        if (t < 8) a_src[n * HEADS + t] = s;
        else       a_dst[n * HEADS + (t - 8)] = s;
    }
}

// ---------------- K2a: edge leaky-relu logits + segment max (conv1) ----------------
__global__ void k_edge1_max(const int* __restrict__ ei, const float* __restrict__ a_src,
                            const float* __restrict__ a_dst, float* __restrict__ raw,
                            unsigned* __restrict__ m1) {
    int e = blockIdx.x * blockDim.x + threadIdx.x;
    if (e >= ET) return;
    int s, d; edge_sd(ei, e, s, d);
#pragma unroll
    for (int j = 0; j < HEADS; ++j) {
        float z = a_src[s * HEADS + j] + a_dst[d * HEADS + j];
        z = z > 0.f ? z : 0.2f * z;                 // leaky_relu
        raw[(size_t)e * HEADS + j] = z;
        atomicMax(&m1[d * HEADS + j], fenc(z));
    }
}

// ---------------- K2b: exp(z - max) and segment sum (conv1) ----------------
__global__ void k_edge1_exp(const int* __restrict__ ei, const unsigned* __restrict__ m1,
                            float* __restrict__ raw, float* __restrict__ denom) {
    int e = blockIdx.x * blockDim.x + threadIdx.x;
    if (e >= ET) return;
    int s, d; edge_sd(ei, e, s, d); (void)s;
#pragma unroll
    for (int j = 0; j < HEADS; ++j) {
        float w = expf(raw[(size_t)e * HEADS + j] - fdec(m1[d * HEADS + j]));
        raw[(size_t)e * HEADS + j] = w;
        atomicAdd(&denom[d * HEADS + j], w);
    }
}

// ---------------- K3: weighted scatter-add aggregation (conv1) ----------------
__global__ void k_agg1(const int* __restrict__ ei, const float* __restrict__ h1,
                       const float* __restrict__ w1e, const float* __restrict__ denom,
                       float* __restrict__ out1) {
    int e = blockIdx.x;
    int t = threadIdx.x;
    int s, d; edge_sd(ei, e, s, d);
#pragma unroll
    for (int j = 0; j < HEADS; ++j) {
        float coef = w1e[(size_t)e * HEADS + j] / (denom[d * HEADS + j] + 1e-16f);
        int col = j * 256 + t;
        atomicAdd(&out1[(size_t)d * F1 + col], coef * h1[(size_t)s * F1 + col]);
    }
}

// ---------------- K4: + bias, ELU, convert to f16 (packed pairs) ----------------
__global__ void k_elu_pack(const float* __restrict__ out1, const float* __restrict__ b1,
                           unsigned* __restrict__ h2) {
    int id = blockIdx.x * blockDim.x + threadIdx.x;    // N*1024 pair slots
    int n = id >> 10, p = id & 1023, col = p * 2;
    float v0 = out1[(size_t)n * F1 + col]     + b1[col];
    float v1 = out1[(size_t)n * F1 + col + 1] + b1[col + 1];
    v0 = v0 > 0.f ? v0 : expf(v0) - 1.f;
    v1 = v1 > 0.f ? v1 : expf(v1) - 1.f;
    union { _Float16 h[2]; unsigned u; } pk;
    pk.h[0] = (_Float16)v0; pk.h[1] = (_Float16)v1;
    h2[(size_t)n * (F1 / 2) + p] = pk.u;
}

// ---------------- K_packB: W2 -> WMMA B-fragment dword layout ----------------
// Bp[((nt*KT2+kt)*8+v)*32+lane] holds B[k0][n],B[k0+1][n] with
// n = nt*16 + lane%16, khalf = lane/16, k0 = kt*32 + khalf*16 + 2v
__global__ void k_packB(const float* __restrict__ W2, unsigned* __restrict__ Bp) {
    int id = blockIdx.x * blockDim.x + threadIdx.x;    // 16*64*8*32 = 262144
    int lane = id & 31, v = (id >> 5) & 7, kt = (id >> 8) & 63, nt = id >> 14;
    int khalf = lane >> 4, nn = nt * 16 + (lane & 15);
    int k0 = kt * 32 + khalf * 16 + 2 * v;
    union { _Float16 h[2]; unsigned u; } pk;
    pk.h[0] = (_Float16)W2[(size_t)k0 * HID + nn];
    pk.h[1] = (_Float16)W2[(size_t)(k0 + 1) * HID + nn];
    Bp[id] = pk.u;
}

// ---------------- K5: WMMA GEMM  g[N,256] = h2[N,2048] @ W2 ----------------
__global__ void __launch_bounds__(512) k_gemm2(const unsigned* __restrict__ h2,
                                               const unsigned* __restrict__ Bp,
                                               float* __restrict__ g) {
    __shared__ __align__(16) _Float16 As[16 * F1];     // 64 KB A stripe
    int rt = blockIdx.x;                               // 1250 row tiles of 16
    int t  = threadIdx.x;                              // 512 threads = 16 waves
    // Stage A stripe: 16 rows x 2048 halfs = 4096 uint4, 8 per thread, coalesced
    const uint4* srcv = (const uint4*)h2;
    uint4* dstv = (uint4*)As;
#pragma unroll
    for (int i = 0; i < 8; ++i) {
        int idx = t + 512 * i;                         // idx = r*256 + c
        int r = idx >> 8, c = idx & 255;
        dstv[idx] = srcv[(size_t)(rt * 16 + r) * 256 + c];
    }
    __syncthreads();

    int w = t >> 5, lane = t & 31;
    int m = lane & 15, khalf = lane >> 4;
    int nt = w;                                        // each wave owns one N tile
    v8f acc = {0.f, 0.f, 0.f, 0.f, 0.f, 0.f, 0.f, 0.f};
    for (int kt = 0; kt < KT2; ++kt) {
        union { unsigned u[8]; v16h h; } a, b;
        // A fragment (contiguous k order, matching Bp): K=16*khalf+2v
#pragma unroll
        for (int v = 0; v < 8; ++v) {
            int bk = kt * 32 + 16 * khalf + 2 * v;
            a.u[v] = *(const unsigned*)&As[m * F1 + bk];
        }
        const unsigned* bp = Bp + ((size_t)(nt * KT2 + kt) * 8) * 32 + lane;
#pragma unroll
        for (int v = 0; v < 8; ++v) b.u[v] = bp[v * 32];
        acc = __builtin_amdgcn_wmma_f32_16x16x32_f16(false, a.h, false, b.h,
                                                     (short)0, acc, false, false);
    }
    // C/D layout: VGPR r -> row = r + 8*khalf, col = lane%16
    int col  = nt * 16 + m;
    int rowb = rt * 16 + 8 * khalf;
#pragma unroll
    for (int r = 0; r < 8; ++r) g[(size_t)(rowb + r) * HID + col] = acc[r];
}

// ---------------- K6: attention logits for conv2 (1 head) ----------------
__global__ void k_att2(const float* __restrict__ g, const float* __restrict__ as2,
                       const float* __restrict__ ad2, float* __restrict__ a_src2,
                       float* __restrict__ a_dst2) {
    int w = threadIdx.x >> 5, lane = threadIdx.x & 31;
    int n = blockIdx.x * 8 + w;
    float s = 0.f, d = 0.f;
#pragma unroll
    for (int j = 0; j < 8; ++j) {
        int c = lane + 32 * j;
        float gv = g[(size_t)n * HID + c];
        s += gv * as2[c];
        d += gv * ad2[c];
    }
    for (int m = 16; m >= 1; m >>= 1) { s += __shfl_xor(s, m, 32); d += __shfl_xor(d, m, 32); }
    if (lane == 0) { a_src2[n] = s; a_dst2[n] = d; }
}

// ---------------- K7/K8: edge softmax passes (conv2, 1 head) ----------------
__global__ void k_edge2_max(const int* __restrict__ ei, const float* __restrict__ a_src,
                            const float* __restrict__ a_dst, float* __restrict__ raw,
                            unsigned* __restrict__ m2) {
    int e = blockIdx.x * blockDim.x + threadIdx.x;
    if (e >= ET) return;
    int s, d; edge_sd(ei, e, s, d);
    float z = a_src[s] + a_dst[d];
    z = z > 0.f ? z : 0.2f * z;
    raw[e] = z;
    atomicMax(&m2[d], fenc(z));
}

__global__ void k_edge2_exp(const int* __restrict__ ei, const unsigned* __restrict__ m2,
                            float* __restrict__ raw, float* __restrict__ denom) {
    int e = blockIdx.x * blockDim.x + threadIdx.x;
    if (e >= ET) return;
    int s, d; edge_sd(ei, e, s, d); (void)s;
    float w = expf(raw[e] - fdec(m2[d]));
    raw[e] = w;
    atomicAdd(&denom[d], w);
}

// ---------------- K_initout: out = broadcast(b2) ----------------
__global__ void k_initout(float* __restrict__ out, const float* __restrict__ b2) {
    int id = blockIdx.x * blockDim.x + threadIdx.x;    // N*256
    out[id] = b2[id & 255];
}

// ---------------- K9: weighted scatter-add aggregation (conv2) ----------------
__global__ void k_agg2(const int* __restrict__ ei, const float* __restrict__ g,
                       const float* __restrict__ w2e, const float* __restrict__ denom,
                       float* __restrict__ out) {
    int e = blockIdx.x;
    int t = threadIdx.x;                               // 256 = HID
    int s, d; edge_sd(ei, e, s, d);
    float coef = w2e[e] / (denom[d] + 1e-16f);
    atomicAdd(&out[(size_t)d * HID + t], coef * g[(size_t)s * HID + t]);
}

// ---------------- host launch ----------------
extern "C" void kernel_launch(void* const* d_in, const int* in_sizes, int n_in,
                              void* d_out, int out_size, void* d_ws, size_t ws_size,
                              hipStream_t stream) {
    (void)in_sizes; (void)n_in; (void)out_size; (void)ws_size;
    const float* x    = (const float*)d_in[0];
    const int*   ei   = (const int*)  d_in[1];   // edge_index (integer)
    const float* W1   = (const float*)d_in[2];
    const float* as1  = (const float*)d_in[3];
    const float* ad1  = (const float*)d_in[4];
    const float* b1   = (const float*)d_in[5];
    const float* W2   = (const float*)d_in[6];
    const float* as2  = (const float*)d_in[7];
    const float* ad2  = (const float*)d_in[8];
    const float* b2   = (const float*)d_in[9];
    float* out = (float*)d_out;

    // bump allocator over workspace (256B aligned)
    char* p = (char*)d_ws;
    auto alloc = [&](size_t bytes) -> void* {
        void* r = (void*)p;
        p += (bytes + 255) & ~(size_t)255;
        return r;
    };
    float*    h1     = (float*)   alloc((size_t)NN * F1 * 4);       // 163.8 MB
    float*    out1   = (float*)   alloc((size_t)NN * F1 * 4);       // 163.8 MB
    unsigned* h2     = (unsigned*)alloc((size_t)NN * F1 * 2);       //  81.9 MB (f16)
    float*    g      = (float*)   alloc((size_t)NN * HID * 4);      //  20.5 MB
    unsigned* Bp     = (unsigned*)alloc((size_t)F1 * HID * 2);      //   1.0 MB
    float*    a_src1 = (float*)   alloc((size_t)NN * HEADS * 4);
    float*    a_dst1 = (float*)   alloc((size_t)NN * HEADS * 4);
    unsigned* m1     = (unsigned*)alloc((size_t)NN * HEADS * 4);
    float*    den1   = (float*)   alloc((size_t)NN * HEADS * 4);
    float*    alpha1 = (float*)   alloc((size_t)ET * HEADS * 4);    //   5.8 MB
    float*    a_src2 = (float*)   alloc((size_t)NN * 4);
    float*    a_dst2 = (float*)   alloc((size_t)NN * 4);
    unsigned* m2     = (unsigned*)alloc((size_t)NN * 4);
    float*    den2   = (float*)   alloc((size_t)NN * 4);
    float*    alpha2 = (float*)   alloc((size_t)ET * 4);

    // zero accumulators (graph-capture-safe stream memsets)
    hipMemsetAsync(out1, 0, (size_t)NN * F1 * 4, stream);
    hipMemsetAsync(m1,   0, (size_t)NN * HEADS * 4, stream);
    hipMemsetAsync(den1, 0, (size_t)NN * HEADS * 4, stream);
    hipMemsetAsync(m2,   0, (size_t)NN * 4, stream);
    hipMemsetAsync(den2, 0, (size_t)NN * 4, stream);

    const int eB = (ET + 255) / 256;

    // ---- conv1 ----
    k_gemm1_att<<<NN, 256, 0, stream>>>(x, W1, as1, ad1, h1, a_src1, a_dst1);
    k_edge1_max<<<eB, 256, 0, stream>>>(ei, a_src1, a_dst1, alpha1, m1);
    k_edge1_exp<<<eB, 256, 0, stream>>>(ei, m1, alpha1, den1);
    k_agg1<<<ET, 256, 0, stream>>>(ei, h1, alpha1, den1, out1);
    k_elu_pack<<<(NN * (F1 / 2)) / 256, 256, 0, stream>>>(out1, b1, h2);

    // ---- conv2 ----
    k_packB<<<(16 * KT2 * 8 * 32) / 256, 256, 0, stream>>>(W2, Bp);
    k_gemm2<<<NN / 16, 512, 0, stream>>>(h2, Bp, g);
    k_att2<<<NN / 8, 256, 0, stream>>>(g, as2, ad2, a_src2, a_dst2);
    k_edge2_max<<<eB, 256, 0, stream>>>(ei, a_src2, a_dst2, alpha2, m2);
    k_edge2_exp<<<eB, 256, 0, stream>>>(ei, m2, alpha2, den2);
    k_initout<<<(NN * HID) / 256, 256, 0, stream>>>(out, b2);
    k_agg2<<<ET, 256, 0, stream>>>(ei, g, alpha2, den2, out);
}
